// UserEncoder_35356170780886
// MI455X (gfx1250) — matrix operand order repaired and checked by reading, not compiled, not run on hardware
//
#include <hip/hip_runtime.h>

// GRU user-encoder, fused single kernel for gfx1250 (MI455X).
// B=1024, T=200, E=H=128, gates=384 (r,z,n). One workgroup = 16 batch rows.
// Wave w owns hidden columns [w*16, w*16+16) across ALL THREE gates: gates and
// hidden state live in registers; W_hh fragments are register-resident.
// Software-pipelined: x_{t+1} gather overlaps step t's GEMM; double-buffered
// x and h-mirror LDS buffers -> ONE barrier per timestep.

#define BATCH 1024
#define SEQ   200
#define EMB   128
#define HID   128
#define GATES 384

#define WP 136               // padded LDS stride for bf16 weight rows (elems)
#define XP 136               // xbuf / hmir stride (bf16 elems)
#define XBUF_SZ (16 * XP)    // one x / h-mirror buffer (elems)

typedef __attribute__((ext_vector_type(16))) __bf16          v16bf;
typedef __attribute__((ext_vector_type(8)))  float           v8f;
typedef __attribute__((ext_vector_type(8)))  unsigned short  u16x8;
typedef __attribute__((ext_vector_type(4)))  float           f32x4;

union Frag {
    v16bf v;
    u16x8 h[2];
    unsigned u32[8];
    unsigned short u[16];
};

// pack two f32 -> two bf16 (truncate) in ONE v_perm_b32
static __device__ __forceinline__ unsigned pack2bf(float lo, float hi) {
    union { float f; unsigned u; } a, b; a.f = hi; b.f = lo;
    return __builtin_amdgcn_perm(a.u, b.u, 0x07060302u);
}

static __device__ __forceinline__ float fast_sigmoid(float x) {
    return __builtin_amdgcn_rcpf(1.0f + __expf(-x));
}
static __device__ __forceinline__ float fast_tanh(float x) {
    float e = __expf(-2.0f * x);
    return 2.0f * __builtin_amdgcn_rcpf(1.0f + e) - 1.0f;
}

// B fragment (32x16): lane l<16 -> N=l, K=kc*32+0..15 ; l>=16 -> N=l-16, K=+16..31
static __device__ __forceinline__ v16bf load_b_frag(const unsigned short* w,
                                                    int n0, int kc, int lane) {
    int col = n0 + (lane & 15);
    int k   = kc * 32 + ((lane & 16) ? 16 : 0);
    const u16x8* p = (const u16x8*)(w + col * WP + k);
    Frag f; f.h[0] = p[0]; f.h[1] = p[1];
    return f.v;
}

// A fragment (16x32) from bf16 LDS: lane l<16 -> M=l, K kc*32+0..15
//                                   lane l>=16 -> M=l-16, K +16..31
static __device__ __forceinline__ v16bf load_a_frag(const unsigned short* x,
                                                    int kc, int lane) {
    int row = lane & 15;
    int k0  = kc * 32 + ((lane & 16) ? 16 : 0);
    const unsigned short* base = x + row * XP + k0;
    Frag f;
    f.h[0] = *(const u16x8*)(base);
    f.h[1] = *(const u16x8*)(base + 8);
    return f.v;
}

extern "C" __global__ void __launch_bounds__(256)
gru_fused_wmma(const int*   __restrict__ item_seq,
               const float* __restrict__ emb,
               const float* __restrict__ W_ih,
               const float* __restrict__ W_hh,
               const float* __restrict__ b_ih,
               const float* __restrict__ b_hh,
               float*       __restrict__ out) {
    extern __shared__ char smem[];
    unsigned short* w_ih = (unsigned short*)smem;     // 384*WP bf16
    unsigned short* xbuf = w_ih + GATES * WP;         // 2 x 16*XP bf16 (double buf)
    unsigned short* hmir = xbuf + 2 * XBUF_SZ;        // 2 x 16*XP bf16 (double buf)

    const int tid   = threadIdx.x;
    const int lane  = tid & 31;
    const int wave  = tid >> 5;        // 8 waves; wave w -> hidden cols [16w,16w+16)
    const int b0    = blockIdx.x * 16; // batch tile base
    const int cl    = lane & 15;
    const int rbase = (lane & 16) ? 8 : 0;

    // ---- one-time: W_ih -> bf16 LDS (pair-packed via v_perm); zero hmir[0] ----
    for (int i = tid; i < GATES * (EMB / 2); i += 256) {
        int g = i >> 6;            // gate row
        int e = (i & 63) << 1;     // even element index
        const float* s = W_ih + (size_t)g * EMB + e;
        *(unsigned*)(w_ih + g * WP + e) = pack2bf(s[0], s[1]);
    }
    for (int i = tid; i < XBUF_SZ; i += 256) hmir[i] = 0;

    // ---- one-time: W_hh fragments -> registers (12 frags, 96 VGPRs) ----
    v16bf whh[3][4];
#pragma unroll
    for (int g = 0; g < 3; ++g) {
        int col = g * 128 + wave * 16 + cl;
        const float* src = W_hh + (size_t)col * HID;
#pragma unroll
        for (int kc = 0; kc < 4; ++kc) {
            const float* s = src + kc * 32 + ((lane & 16) ? 16 : 0);
            Frag f;
#pragma unroll
            for (int j = 0; j < 8; ++j) f.u32[j] = pack2bf(s[2 * j], s[2 * j + 1]);
            whh[g][kc] = f.v;
        }
    }

    // per-lane gate biases (column = wave*16 + cl)
    const int gc  = wave * 16 + cl;
    const float brs = b_ih[gc]       + b_hh[gc];
    const float bzs = b_ih[128 + gc] + b_hh[128 + gc];
    const float bni = b_ih[256 + gc];
    const float bnh = b_hh[256 + gc];

    float hreg[8];
#pragma unroll
    for (int v = 0; v < 8; ++v) hreg[v] = 0.0f;

    const int grow = tid >> 4;          // gather: batch row 0..15
    const int ge0  = (tid & 15) << 3;   // gather: 8 consecutive E elems

    // ---- prologue: gather x_0 into xbuf[0] ----
    {
        int item = item_seq[(b0 + grow) * SEQ + 0];
        const float* src = emb + (size_t)item * EMB + ge0;
        f32x4 a = *(const f32x4*)src;
        f32x4 b = *(const f32x4*)(src + 4);
        unsigned* dst = (unsigned*)(xbuf + grow * XP + ge0);
        dst[0] = pack2bf(a.x, a.y);
        dst[1] = pack2bf(a.z, a.w);
        dst[2] = pack2bf(b.x, b.y);
        dst[3] = pack2bf(b.z, b.w);
    }
    __syncthreads();

    for (int t = 0; t < SEQ; ++t) {
        const int cur = t & 1;
        const int nxt = cur ^ 1;
        unsigned short* xcur = xbuf + cur * XBUF_SZ;
        unsigned short* xnxt = xbuf + nxt * XBUF_SZ;
        unsigned short* hcur = hmir + cur * XBUF_SZ;
        unsigned short* hnxt = hmir + nxt * XBUF_SZ;

        // ---- stage A: issue x_{t+1} global loads (overlap with GEMM below) ----
        const bool have_next = (t + 1 < SEQ);
        f32x4 ga = {0.f, 0.f, 0.f, 0.f}, gb = {0.f, 0.f, 0.f, 0.f};
        if (have_next) {
            int item = item_seq[(b0 + grow) * SEQ + t + 1];
            const float* src = emb + (size_t)item * EMB + ge0;
            ga = *(const f32x4*)src;
            gb = *(const f32x4*)(src + 4);
        }
        if (t + 2 < SEQ) {   // keep t+2 rows warm in L2
            int p = item_seq[(b0 + grow) * SEQ + t + 2];
            __builtin_prefetch(emb + (size_t)p * EMB + ge0, 0, 0);
        }

        // ---- stage B: aci = x_t @ W_ih^T ; ach = h_t @ W_hh^T (WMMA bf16) ----
        // hidden-side split into two partial chains to halve dependency depth
        v8f aci[3]  = {{0}, {0}, {0}};
        v8f ach0[3] = {{0}, {0}, {0}};
        v8f ach1[3] = {{0}, {0}, {0}};
#pragma unroll
        for (int kc = 0; kc < 4; ++kc) {
            v16bf ax = load_a_frag(xcur, kc, lane);
            v16bf ah = load_a_frag(hcur, kc, lane);
#pragma unroll
            for (int g = 0; g < 3; ++g) {
                v16bf bi = load_b_frag(w_ih, (g * 8 + wave) * 16, kc, lane);
                aci[g] = __builtin_amdgcn_wmma_f32_16x16x32_bf16(
                    false, ax, false, bi, (short)0, aci[g], false, false);
                if (kc < 2)
                    ach0[g] = __builtin_amdgcn_wmma_f32_16x16x32_bf16(
                        false, ah, false, whh[g][kc], (short)0, ach0[g], false, false);
                else
                    ach1[g] = __builtin_amdgcn_wmma_f32_16x16x32_bf16(
                        false, ah, false, whh[g][kc], (short)0, ach1[g], false, false);
            }
        }

        // ---- stage C: commit x_{t+1} to the other x buffer ----
        if (have_next) {
            unsigned* dst = (unsigned*)(xnxt + grow * XP + ge0);
            dst[0] = pack2bf(ga.x, ga.y);
            dst[1] = pack2bf(ga.z, ga.w);
            dst[2] = pack2bf(gb.x, gb.y);
            dst[3] = pack2bf(gb.z, gb.w);
        }

        // ---- stage D: gates + h update in registers; bf16 mirror -> hnxt ----
#pragma unroll
        for (int v = 0; v < 8; ++v) {
            float pr = aci[0][v] + ach0[0][v] + ach1[0][v] + brs;
            float pz = aci[1][v] + ach0[1][v] + ach1[1][v] + bzs;
            float xn = aci[2][v] + bni;
            float hn = ach0[2][v] + ach1[2][v] + bnh;
            float r = fast_sigmoid(pr);
            float z = fast_sigmoid(pz);
            float n = fast_tanh(xn + r * hn);
            float hnew = n + z * (hreg[v] - n);
            hreg[v] = hnew;
            hnxt[(rbase + v) * XP + gc] =
                (unsigned short)(__builtin_bit_cast(unsigned, hnew) >> 16);
        }

        // one barrier per timestep: publishes x_{t+1} and h_{t+1}
        __syncthreads();
    }

    // ---- final hidden state (registers) -> out ----
#pragma unroll
    for (int v = 0; v < 8; ++v)
        out[(size_t)(b0 + rbase + v) * HID + gc] = hreg[v];
}

extern "C" void kernel_launch(void* const* d_in, const int* in_sizes, int n_in,
                              void* d_out, int out_size, void* d_ws, size_t ws_size,
                              hipStream_t stream) {
    (void)in_sizes; (void)n_in; (void)out_size; (void)d_ws; (void)ws_size;
    const int*   item_seq = (const int*)d_in[0];
    const float* emb      = (const float*)d_in[1];
    const float* W_ih     = (const float*)d_in[2];
    const float* W_hh     = (const float*)d_in[3];
    const float* b_ih     = (const float*)d_in[4];
    const float* b_hh     = (const float*)d_in[5];

    size_t lds = (size_t)(GATES * WP * sizeof(unsigned short)      // W_ih bf16
                          + 2 * XBUF_SZ * sizeof(unsigned short)   // x double buf
                          + 2 * XBUF_SZ * sizeof(unsigned short)); // h double buf

    gru_fused_wmma<<<dim3(BATCH / 16), dim3(256), lds, stream>>>(
        item_seq, emb, W_ih, W_hh, b_ih, b_hh, (float*)d_out);
}
